// FEDformer_27169963114635
// MI455X (gfx1250) — matrix operand order repaired and checked by reading, not compiled, not run on hardware
//
#include <hip/hip_runtime.h>
#include <hip/hip_bf16.h>
#include <math.h>

// ---------------- model dims ----------------
constexpr int Bsz = 16;
constexpr int LE  = 1024;
constexpr int LD  = 768;
constexpr int D   = 512;
constexpr int H   = 8;
constexpr int E   = 64;    // d_head
constexpr int MM  = 64;    // retained modes
constexpr int CO  = 512;
constexpr int DFF = 1024;

typedef __attribute__((ext_vector_type(16))) _Float16 v16h;
typedef __attribute__((ext_vector_type(8)))  _Float16 v8h;
typedef __attribute__((ext_vector_type(8)))  float    v8f;

// ---------------- workspace layout (floats) ----------------
constexpr size_t SZ_BLED = (size_t)Bsz*LE*D;      // 8,388,608
constexpr size_t SZ_BLDD = (size_t)Bsz*LD*D;      // 6,291,456
constexpr size_t SZ_FREQ = (size_t)Bsz*H*E*2*MM;  // 1,048,576
constexpr size_t SZ_XCAT = (size_t)Bsz*LD*3*D;    // 18,874,368

constexpr size_t WS_A    = 0;
constexpr size_t WS_B    = WS_A    + SZ_BLED;
constexpr size_t WS_C    = WS_B    + SZ_BLED;
constexpr size_t WS_Dv   = WS_C    + SZ_BLED;     // ffn hidden / Xcat
constexpr size_t WS_E    = WS_Dv   + SZ_XCAT;     // encoder output (cross)
constexpr size_t WS_F    = WS_E    + SZ_BLED;     // decoder x
constexpr size_t WS_G    = WS_F    + SZ_BLDD;     // trend sum
constexpr size_t WS_QF   = WS_G    + SZ_BLDD;
constexpr size_t WS_KF   = WS_QF   + SZ_FREQ;
constexpr size_t WS_SEL  = WS_KF   + SZ_FREQ;
constexpr size_t WS_QK   = WS_SEL  + SZ_FREQ;
constexpr size_t WS_QKV  = WS_QK   + SZ_FREQ;
constexpr size_t WS_DFT1024  = WS_QKV + SZ_FREQ;
constexpr size_t WS_IDFT1024 = WS_DFT1024 + (size_t)1024*2*MM;
constexpr size_t WS_DFT768   = WS_IDFT1024 + (size_t)2*MM*1024;
constexpr size_t WS_IDFT768  = WS_DFT768 + (size_t)768*2*MM;
constexpr size_t WS_WCAT     = WS_IDFT768 + (size_t)2*MM*768;

// =====================================================================
// Tiled GEMM: C[M,N] = A[M,K] @ B[K,N] (+bias)(+resid)(relu), f32 in/out,
// f16 WMMA compute (v_wmma_f32_16x16x32_f16), f32 accumulate.
// Block = 256 thr = 8 wave32, tile 128x128, wave tile 32x64 (2x4 WMMA),
// double-buffered LDS (one barrier per 32-K step).
// Requires M%128==0, N%128==0, K%32==0 (true for all call sites).
// =====================================================================
__global__ __launch_bounds__(256)
void gemm_wmma_f16(const float* __restrict__ A, const float* __restrict__ Bm,
                   const float* __restrict__ bias, const float* __restrict__ resid,
                   float* __restrict__ C, int Mrows, int N, int Kdim, int relu)
{
    __shared__ __align__(32) _Float16 As[2][128][48];
    __shared__ __align__(32) _Float16 Bs[2][128][48];

    const int tid  = threadIdx.x;
    const int bm   = blockIdx.y * 128;
    const int bn   = blockIdx.x * 128;
    const int wave = tid >> 5, lane = tid & 31;
    const int wm = wave & 3, wn = wave >> 2;          // 4x2 wave grid
    const int rbase = bm + wm * 32, cbase = bn + wn * 64;
    const int m16 = lane & 15, khalf = lane >> 4;

    v8f acc[2][4] = {};

    // stage one 32-K slab (f32 -> f16), B stored transposed Bs[n][k]
    auto stage = [&](int k0, int buf) {
        #pragma unroll
        for (int i = 0; i < 4; ++i) {
            int v = tid + i * 256;            // 0..1023 float4 slots of A (128x32)
            int r = v >> 3;
            int kc = (v & 7) << 2;
            const float4 f = *(const float4*)(A + (bm + r) * Kdim + k0 + kc);
            As[buf][r][kc + 0] = (_Float16)f.x; As[buf][r][kc + 1] = (_Float16)f.y;
            As[buf][r][kc + 2] = (_Float16)f.z; As[buf][r][kc + 3] = (_Float16)f.w;
        }
        #pragma unroll
        for (int i = 0; i < 4; ++i) {
            int v = tid + i * 256;            // 0..1023 float4 slots of B (32x128)
            int kr = v >> 5;                  // 0..31
            int nc = (v & 31) << 2;
            const float4 f = *(const float4*)(Bm + (k0 + kr) * N + bn + nc);
            Bs[buf][nc + 0][kr] = (_Float16)f.x; Bs[buf][nc + 1][kr] = (_Float16)f.y;
            Bs[buf][nc + 2][kr] = (_Float16)f.z; Bs[buf][nc + 3][kr] = (_Float16)f.w;
        }
    };

    stage(0, 0);
    __syncthreads();

    for (int k0 = 0; k0 < Kdim; k0 += 32) {
        const int cur = (k0 >> 5) & 1;
        if (k0 + 32 < Kdim) {
            if (k0 + 64 < Kdim)  // keep one slab ahead warming L2 (global_prefetch_b8)
                __builtin_prefetch(A + (bm + (tid >> 1)) * Kdim + k0 + 64, 0, 1);
            stage(k0 + 32, cur ^ 1);          // overlap with WMMA below
        }

        // fragments in contiguous K order, same for A and B (16-bit A 16x32; B 32x16)
        v16h a[2], b[4];
        #pragma unroll
        for (int i = 0; i < 2; ++i) {
            const _Float16* ap = &As[cur][wm * 32 + i * 16 + m16][0];
            a[i] = *(const v16h*)(ap + khalf * 16);         // K = khalf*16 + 0..15
        }
        #pragma unroll
        for (int j = 0; j < 4; ++j) {
            const _Float16* bp = &Bs[cur][wn * 64 + j * 16 + m16][0];
            b[j] = *(const v16h*)(bp + khalf * 16);         // K = khalf*16 + 0..15
        }
        #pragma unroll
        for (int i = 0; i < 2; ++i)
            #pragma unroll
            for (int j = 0; j < 4; ++j)
                acc[i][j] = __builtin_amdgcn_wmma_f32_16x16x32_f16(
                    false, a[i], false, b[j], (short)0, acc[i][j], false, false);
        __syncthreads();
    }

    // epilogue per C/D layout: VGPR r, lanes<16 -> M=r, lanes>=16 -> M=8+r
    const int col0 = lane & 15;
    const int rsel = (lane >> 4) * 8;
    #pragma unroll
    for (int i = 0; i < 2; ++i)
        #pragma unroll
        for (int j = 0; j < 4; ++j) {
            int cc = cbase + j * 16 + col0;
            float bv = bias ? bias[cc] : 0.f;
            #pragma unroll
            for (int r = 0; r < 8; ++r) {
                int rr = rbase + i * 16 + rsel + r;
                int off = rr * N + cc;                       // fits 31 bits everywhere
                float v = acc[i][j][r] + bv;
                if (resid) v += resid[off];
                if (relu)  v = fmaxf(v, 0.f);
                C[off] = v;
            }
        }
}

// ---------------- [B,L,D] -> [B,H,E,L] tiled transpose ----------------
__global__ void k_to_heads(const float* __restrict__ in, float* __restrict__ out, int L)
{
    __shared__ float t[32][33];
    int bh = blockIdx.z; int b = bh >> 3; int h = bh & 7;
    int l0 = blockIdx.x * 32, e0 = blockIdx.y * 32;
    int tx = threadIdx.x, ty = threadIdx.y;
    const float* ip = in + (size_t)b * L * D + h * E;
    float* op = out + (size_t)bh * E * L;
    #pragma unroll
    for (int j = 0; j < 32; j += 8) t[ty + j][tx] = ip[(size_t)(l0 + ty + j) * D + e0 + tx];
    __syncthreads();
    #pragma unroll
    for (int j = 0; j < 32; j += 8) op[(size_t)(e0 + ty + j) * L + l0 + tx] = t[tx][ty + j];
}

// ---------------- [B,H,E,L] -> [B,L,D] ----------------
__global__ void k_from_heads(const float* __restrict__ in, float* __restrict__ out, int L)
{
    __shared__ float t[32][33];
    int bh = blockIdx.z; int b = bh >> 3; int h = bh & 7;
    int l0 = blockIdx.x * 32, e0 = blockIdx.y * 32;
    int tx = threadIdx.x, ty = threadIdx.y;
    const float* ip = in + (size_t)bh * E * L;
    float* op = out + (size_t)b * L * D + h * E;
    #pragma unroll
    for (int j = 0; j < 32; j += 8) t[ty + j][tx] = ip[(size_t)(e0 + ty + j) * L + l0 + tx];
    __syncthreads();
    #pragma unroll
    for (int j = 0; j < 32; j += 8) op[(size_t)(l0 + ty + j) * D + e0 + tx] = t[tx][ty + j];
}

// ---------------- truncated DFT / inverse-DFT basis ----------------
// dft [L, 2M]: cos(2πml/L), -sin(...);  idft [2M, L]: c_m cos, -c_m sin
__global__ void k_gen_dft(float* __restrict__ dft, float* __restrict__ idft, int L)
{
    int t = blockIdx.x * blockDim.x + threadIdx.x;
    if (t >= L * MM) return;
    int l = t / MM, m = t % MM;
    int r = (m * l) % L;                       // exact angle reduction
    float th = 6.28318530717958647692f * (float)r / (float)L;
    float s, c; __sincosf(th, &s, &c);
    dft[(size_t)l * (2 * MM) + m]      = c;
    dft[(size_t)l * (2 * MM) + MM + m] = -s;
    float cm = (m == 0 ? 1.0f : 2.0f) / (float)L;
    idft[(size_t)m * L + l]            = cm * c;
    idft[(size_t)(MM + m) * L + l]     = -cm * s;
}

// ---------------- per-mode complex E x E mixing ----------------
// S[b,h,o,m] = scale * sum_i Q[b,h,i,m] * (wr+i·wi)[h,i,o,m]
__global__ void k_modemix(const float* __restrict__ Q, const float* __restrict__ wr,
                          const float* __restrict__ wi, float* __restrict__ S, float scale)
{
    int t = blockIdx.x * blockDim.x + threadIdx.x;
    if (t >= Bsz * H * E * MM) return;
    int m = t % MM, o = (t / MM) % E, bh = t / (MM * E), h = bh & 7;
    const float* qb  = Q + (size_t)bh * E * 2 * MM;
    const float* wrb = wr + ((size_t)h * E * E + o) * MM + m;
    const float* wib = wi + ((size_t)h * E * E + o) * MM + m;
    float re = 0.f, im = 0.f;
    for (int i = 0; i < E; ++i) {
        float qr = qb[(size_t)i * 2 * MM + m];
        float qi = qb[(size_t)i * 2 * MM + MM + m];
        float a  = wrb[(size_t)i * E * MM];
        float bb = wib[(size_t)i * E * MM];
        re += qr * a - qi * bb;
        im += qr * bb + qi * a;
    }
    float* sp = S + (size_t)bh * E * 2 * MM + (size_t)o * 2 * MM;
    sp[m]      = re * scale;
    sp[MM + m] = im * scale;
}

// ---------------- cross: qk = complex-tanh(qf · kf over e) ----------------
__global__ void k_qk_tanh(const float* __restrict__ QF, const float* __restrict__ KF,
                          float* __restrict__ QK)
{
    int t = blockIdx.x * blockDim.x + threadIdx.x;
    if (t >= Bsz * H * MM * MM) return;
    int y = t % MM, x = (t / MM) % MM, bh = t / (MM * MM);
    const float* qb = QF + (size_t)bh * E * 2 * MM;
    const float* kb = KF + (size_t)bh * E * 2 * MM;
    float re = 0.f, im = 0.f;
    for (int e = 0; e < E; ++e) {
        float qr = qb[(size_t)e * 2 * MM + x], qi = qb[(size_t)e * 2 * MM + MM + x];
        float kr = kb[(size_t)e * 2 * MM + y], ki = kb[(size_t)e * 2 * MM + MM + y];
        re += qr * kr - qi * ki;
        im += qr * ki + qi * kr;
    }
    // tanh(a+bi) = (sinh 2a + i sin 2b) / (cosh 2a + cos 2b), saturated for big |a|
    float tr = 2.f * re, ti = 2.f * im, outr, outi;
    if (fabsf(tr) > 20.f) { outr = tr > 0.f ? 1.f : -1.f; outi = 0.f; }
    else {
        float sh = sinhf(tr), ch = coshf(tr), sn, cs;
        __sincosf(ti, &sn, &cs);
        float d = ch + cs;
        outr = sh / d; outi = sn / d;
    }
    float* op = QK + (size_t)bh * MM * 2 * MM + (size_t)x * 2 * MM;
    op[y] = outr; op[MM + y] = outi;
}

// ---------------- cross: qkv[b,h,e,x] = sum_y qk[x,y] * kf[e,y] ----------------
__global__ void k_qkv(const float* __restrict__ QK, const float* __restrict__ KF,
                      float* __restrict__ QKV)
{
    int t = blockIdx.x * blockDim.x + threadIdx.x;
    if (t >= Bsz * H * E * MM) return;
    int x = t % MM, e = (t / MM) % E, bh = t / (MM * E);
    const float* qb = QK + (size_t)bh * MM * 2 * MM + (size_t)x * 2 * MM;
    const float* kb = KF + (size_t)bh * E * 2 * MM + (size_t)e * 2 * MM;
    float re = 0.f, im = 0.f;
    for (int y = 0; y < MM; ++y) {
        float ar = qb[y], ai = qb[MM + y];
        float br = kb[y], bi = kb[MM + y];
        re += ar * br - ai * bi;
        im += ar * bi + ai * br;
    }
    float* op = QKV + (size_t)bh * E * 2 * MM + (size_t)e * 2 * MM;
    op[x] = re; op[MM + x] = im;
}

// ---------------- series_decomp: sliding 25-tap replicate-pad mean ----------------
// in-place seasonal; trend: tmode 0=skip, 1=store, 2=accumulate
__global__ void k_decomp(float* __restrict__ X, float* __restrict__ T, int L, int tmode)
{
    int t = blockIdx.x * blockDim.x + threadIdx.x;
    if (t >= Bsz * D) return;
    int b = t / D, c = t % D;
    float* xp = X + (size_t)b * L * D + c;
    float* tp = T ? T + (size_t)b * L * D + c : nullptr;
    float ring[26];
    for (int j = 0; j <= 12; ++j) ring[j] = xp[(size_t)j * D];
    float x0 = ring[0];
    float xlast = xp[(size_t)(L - 1) * D];
    float sum = 13.f * x0;
    for (int j = 1; j <= 12; ++j) sum += ring[j];
    for (int l = 0; l < L; ++l) {
        float mean = sum * (1.f / 25.f);
        float xv = ring[l % 26];
        xp[(size_t)l * D] = xv - mean;
        if (tmode == 1)      tp[(size_t)l * D] = mean;
        else if (tmode == 2) tp[(size_t)l * D] += mean;
        int ja = l + 13;
        float addv = (ja < L) ? (ring[ja % 26] = xp[(size_t)ja * D]) : xlast;
        int js = l - 12;
        float subv = (js <= 0) ? x0 : ring[js % 26];
        sum += addv - subv;
    }
}

// ---------------- channel LayerNorm per row (D=512, 256 threads) ----------------
__global__ __launch_bounds__(256)
void k_lnrow(const float* __restrict__ X, float* __restrict__ Y,
             const float* __restrict__ g, const float* __restrict__ bb)
{
    __shared__ float red[256];
    size_t row = blockIdx.x;
    const float* xp = X + row * D;
    float* yp = Y + row * D;
    int tid = threadIdx.x;
    float v0 = xp[tid], v1 = xp[tid + 256];
    red[tid] = v0 + v1; __syncthreads();
    for (int s = 128; s > 0; s >>= 1) { if (tid < s) red[tid] += red[tid + s]; __syncthreads(); }
    float mean = red[0] * (1.f / 512.f);
    __syncthreads();
    float d0 = v0 - mean, d1 = v1 - mean;
    red[tid] = d0 * d0 + d1 * d1; __syncthreads();
    for (int s = 128; s > 0; s >>= 1) { if (tid < s) red[tid] += red[tid + s]; __syncthreads(); }
    float rstd = rsqrtf(red[0] * (1.f / 512.f) + 1e-5f);
    yp[tid]       = g[tid] * d0 * rstd + bb[tid];
    yp[tid + 256] = g[tid + 256] * d1 * rstd + bb[tid + 256];
}

// ---------------- subtract per-(b,c) time mean ----------------
__global__ void k_subtmean(float* __restrict__ X, int L)
{
    int t = blockIdx.x * blockDim.x + threadIdx.x;
    if (t >= Bsz * D) return;
    int b = t / D, c = t % D;
    float* xp = X + (size_t)b * L * D + c;
    float s = 0.f;
    for (int l = 0; l < L; ++l) s += xp[(size_t)l * D];
    float m = s / (float)L;
    for (int l = 0; l < L; ++l) xp[(size_t)l * D] -= m;
}

// ---------------- conv weight rearrange: Wcat[tap*D+ci][co] = w[co][ci][tap] ----------------
__global__ void k_wcat(const float* __restrict__ w, float* __restrict__ Wcat)
{
    int t = blockIdx.x * blockDim.x + threadIdx.x;
    if (t >= 3 * D * CO) return;
    int co = t % CO, r = t / CO, ci = r % D, tap = r / D;
    Wcat[t] = w[((size_t)co * D + ci) * 3 + tap];
}

// ---------------- circular-shift im2col: Xcat[(b,l)][tap*D+ci] = ts[b][(l-1+tap)%L][ci] ----
__global__ void k_xcat(const float* __restrict__ TS, float* __restrict__ XC)
{
    int t = blockIdx.x * blockDim.x + threadIdx.x;
    if (t >= Bsz * LD * 3 * D) return;
    int col = t % (3 * D), row = t / (3 * D);
    int b = row / LD, l = row % LD;
    int tap = col / D, ci = col % D;
    int sl = (l - 1 + tap + LD) % LD;
    XC[t] = TS[((size_t)b * LD + sl) * D + ci];
}

// =====================================================================
// host orchestration
// =====================================================================
static inline void gemm(hipStream_t s, const float* A, const float* B, const float* bias,
                        const float* resid, float* C, int Mr, int N, int K, int relu = 0)
{
    dim3 g((unsigned)(N / 128), (unsigned)(Mr / 128));
    gemm_wmma_f16<<<g, 256, 0, s>>>(A, B, bias, resid, C, Mr, N, K, relu);
}

extern "C" void kernel_launch(void* const* d_in, const int* in_sizes, int n_in,
                              void* d_out, int out_size, void* d_ws, size_t ws_size,
                              hipStream_t stream)
{
    (void)in_sizes; (void)n_in; (void)out_size; (void)ws_size;
    auto IN = [&](int i) { return (const float*)d_in[i]; };
    float* ws = (float*)d_ws;
    float* Abuf = ws + WS_A;   float* Bbuf = ws + WS_B;   float* Cbuf = ws + WS_C;
    float* Dbuf = ws + WS_Dv;  float* Ebuf = ws + WS_E;   float* Fbuf = ws + WS_F;
    float* Gbuf = ws + WS_G;
    float* QF  = ws + WS_QF;   float* KF  = ws + WS_KF;   float* SEL = ws + WS_SEL;
    float* QKb = ws + WS_QK;   float* QKV = ws + WS_QKV;
    float* DFT1024 = ws + WS_DFT1024;  float* IDFT1024 = ws + WS_IDFT1024;
    float* DFT768  = ws + WS_DFT768;   float* IDFT768  = ws + WS_IDFT768;
    float* WCAT    = ws + WS_WCAT;

    const int NBHE = Bsz * H * E;            // 8192 DFT rows
    const int MIXT = Bsz * H * E * MM;       // 524288

    // DFT bases (deterministic, regenerated every call)
    k_gen_dft<<<(1024 * MM) / 256, 256, 0, stream>>>(DFT1024, IDFT1024, 1024);
    k_gen_dft<<<(768 * MM) / 256, 256, 0, stream>>>(DFT768, IDFT768, 768);

    // ---------------- encoder: 2 layers ----------------
    const float* xin = IN(0);
    for (int p = 0; p < 2; ++p) {
        int pb = 3 + p * 12;   // wq bq wk bk wv bv wo bo fwr fwi w1 w2
        // self frequency attention (only Q path is used by fourier_block)
        gemm(stream, xin, IN(pb + 0), IN(pb + 1), nullptr, Bbuf, Bsz * LE, D, D);
        k_to_heads<<<dim3(LE / 32, 2, Bsz * H), dim3(32, 8), 0, stream>>>(Bbuf, Cbuf, LE);
        gemm(stream, Cbuf, DFT1024, nullptr, nullptr, QF, NBHE, 2 * MM, LE);
        k_modemix<<<MIXT / 256, 256, 0, stream>>>(QF, IN(pb + 8), IN(pb + 9), SEL, 1.0f);
        gemm(stream, SEL, IDFT1024, nullptr, nullptr, Cbuf, NBHE, LE, 2 * MM);
        k_from_heads<<<dim3(LE / 32, 2, Bsz * H), dim3(32, 8), 0, stream>>>(Cbuf, Bbuf, LE);
        gemm(stream, Bbuf, IN(pb + 6), IN(pb + 7), xin, Abuf, Bsz * LE, D, D);
        k_decomp<<<(Bsz * D) / 256, 256, 0, stream>>>(Abuf, nullptr, LE, 0);
        // FFN (no biases in reference ffn)
        gemm(stream, Abuf, IN(pb + 10), nullptr, nullptr, Dbuf, Bsz * LE, DFF, D, 1);
        gemm(stream, Dbuf, IN(pb + 11), nullptr, Abuf, Abuf, Bsz * LE, D, DFF);
        k_decomp<<<(Bsz * D) / 256, 256, 0, stream>>>(Abuf, nullptr, LE, 0);
        xin = Abuf;
    }
    // encoder tlayernorm -> cross context
    k_lnrow<<<Bsz * LE, 256, 0, stream>>>(Abuf, Ebuf, IN(27), IN(28));
    k_subtmean<<<(Bsz * D) / 256, 256, 0, stream>>>(Ebuf, LE);

    // ---------------- decoder ----------------
    {   // self attention (params 29..38)
        int pb = 29;
        gemm(stream, IN(1), IN(pb + 0), IN(pb + 1), nullptr, Bbuf, Bsz * LD, D, D);
        k_to_heads<<<dim3(LD / 32, 2, Bsz * H), dim3(32, 8), 0, stream>>>(Bbuf, Cbuf, LD);
        gemm(stream, Cbuf, DFT768, nullptr, nullptr, QF, NBHE, 2 * MM, LD);
        k_modemix<<<MIXT / 256, 256, 0, stream>>>(QF, IN(pb + 8), IN(pb + 9), SEL, 1.0f);
        gemm(stream, SEL, IDFT768, nullptr, nullptr, Cbuf, NBHE, LD, 2 * MM);
        k_from_heads<<<dim3(LD / 32, 2, Bsz * H), dim3(32, 8), 0, stream>>>(Cbuf, Bbuf, LD);
        gemm(stream, Bbuf, IN(pb + 6), IN(pb + 7), IN(1), Fbuf, Bsz * LD, D, D);
    }
    k_decomp<<<(Bsz * D) / 256, 256, 0, stream>>>(Fbuf, Gbuf, LD, 1);   // t1

    {   // cross frequency attention (params 39..48)
        int pb = 39;
        gemm(stream, Fbuf, IN(pb + 0), IN(pb + 1), nullptr, Bbuf, Bsz * LD, D, D); // q
        k_to_heads<<<dim3(LD / 32, 2, Bsz * H), dim3(32, 8), 0, stream>>>(Bbuf, Cbuf, LD);
        gemm(stream, Cbuf, DFT768, nullptr, nullptr, QF, NBHE, 2 * MM, LD);
        gemm(stream, Ebuf, IN(pb + 2), IN(pb + 3), nullptr, Bbuf, Bsz * LE, D, D); // k
        k_to_heads<<<dim3(LE / 32, 2, Bsz * H), dim3(32, 8), 0, stream>>>(Bbuf, Cbuf, LE);
        gemm(stream, Cbuf, DFT1024, nullptr, nullptr, KF, NBHE, 2 * MM, LE);
        k_qk_tanh<<<(Bsz * H * MM * MM) / 256, 256, 0, stream>>>(QF, KF, QKb);
        k_qkv<<<MIXT / 256, 256, 0, stream>>>(QKb, KF, QKV);
        k_modemix<<<MIXT / 256, 256, 0, stream>>>(QKV, IN(pb + 8), IN(pb + 9), SEL,
                                                  1.0f / (float)(E * E));
        gemm(stream, SEL, IDFT768, nullptr, nullptr, Cbuf, NBHE, LD, 2 * MM);
        k_from_heads<<<dim3(LD / 32, 2, Bsz * H), dim3(32, 8), 0, stream>>>(Cbuf, Bbuf, LD);
        gemm(stream, Bbuf, IN(pb + 6), IN(pb + 7), Fbuf, Fbuf, Bsz * LD, D, D);
    }
    k_decomp<<<(Bsz * D) / 256, 256, 0, stream>>>(Fbuf, Gbuf, LD, 2);   // += t2

    // decoder FFN (49,50)
    gemm(stream, Fbuf, IN(49), nullptr, nullptr, Dbuf, Bsz * LD, DFF, D, 1);
    gemm(stream, Dbuf, IN(50), nullptr, Fbuf, Fbuf, Bsz * LD, D, DFF);
    k_decomp<<<(Bsz * D) / 256, 256, 0, stream>>>(Fbuf, Gbuf, LD, 2);   // += t3

    // trend path: circular conv as im2col GEMM, residual = trend_init -> d_out
    k_wcat<<<(3 * D * CO) / 256, 256, 0, stream>>>(IN(51), WCAT);
    k_xcat<<<(Bsz * LD * 3 * D) / 256, 256, 0, stream>>>(Gbuf, Dbuf);
    gemm(stream, Dbuf, WCAT, nullptr, IN(2), (float*)d_out, Bsz * LD, CO, 3 * D);

    // seasonal path: tlayernorm then projection, accumulated into d_out
    k_lnrow<<<Bsz * LD, 256, 0, stream>>>(Fbuf, Abuf, IN(52), IN(53));
    k_subtmean<<<(Bsz * D) / 256, 256, 0, stream>>>(Abuf, LD);
    gemm(stream, Abuf, IN(54), IN(55), (const float*)d_out, (float*)d_out, Bsz * LD, CO, D);
}
